// MHA_5377299054964
// MI455X (gfx1250) — matrix operand order repaired and checked by reading, not compiled, not run on hardware
//
#include <hip/hip_runtime.h>
#include <cstdint>
#include <cstddef>

// ---------------- problem constants ----------------
#define B_      2
#define S_      2048
#define DMODEL  5120
#define H_      32
#define HKV_    8
#define HD_     128
#define NTOK    (B_ * S_)        // 4096
#define QDIM    (H_ * HD_)       // 4096
#define KVDIM   (HKV_ * HD_)     // 1024
#define SCALE_  0.08838834764831845f   // 128^-0.5

// ---------------- WMMA types ----------------
typedef __attribute__((ext_vector_type(16))) __bf16 bf16x16;
typedef __attribute__((ext_vector_type(8)))  float  f32x8;

union FragA { bf16x16 v; uint4 q[2]; };

__device__ __forceinline__ unsigned short f2bf(float f) {
  unsigned u = __builtin_bit_cast(unsigned, f);
  u += 0x7FFFu + ((u >> 16) & 1u);            // round-to-nearest-even
  return (unsigned short)(u >> 16);
}
__device__ __forceinline__ float bf2f(unsigned short s) {
  unsigned u = ((unsigned)s) << 16;
  return __builtin_bit_cast(float, u);
}

__device__ __forceinline__ f32x8 wmma_bf16(const bf16x16& a, const bf16x16& b, const f32x8& c) {
  return __builtin_amdgcn_wmma_f32_16x16x32_bf16(false, a, false, b, (short)0, c, false, false);
}

// ---------------- DPP16 16-lane-group reductions (no LDS traffic) ----------------
template <int CTRL>
__device__ __forceinline__ float dppf(float x) {
  int y = __builtin_amdgcn_mov_dpp(__builtin_bit_cast(int, x), CTRL, 0xF, 0xF, true);
  return __builtin_bit_cast(float, y);
}
__device__ __forceinline__ float row16_max(float x) {
  x = fmaxf(x, dppf<0xB1>(x));    // quad_perm [1,0,3,2]  : xor 1
  x = fmaxf(x, dppf<0x4E>(x));    // quad_perm [2,3,0,1]  : xor 2
  x = fmaxf(x, dppf<0x141>(x));   // row_half_mirror      : xor 4
  x = fmaxf(x, dppf<0x140>(x));   // row_mirror           : xor 8
  return x;
}
__device__ __forceinline__ float row16_sum(float x) {
  x += dppf<0xB1>(x);
  x += dppf<0x4E>(x);
  x += dppf<0x141>(x);
  x += dppf<0x140>(x);
  return x;
}

// ---------------- async global -> LDS (CDNA5 ASYNCcnt path) ----------------
__device__ __forceinline__ void async_load_b128(unsigned lds_off, const void* gaddr) {
  asm volatile("global_load_async_to_lds_b128 %0, %1, off"
               :: "v"(lds_off), "v"((unsigned long long)gaddr)
               : "memory");
}
__device__ __forceinline__ void wait_async0() {
  asm volatile("s_wait_asynccnt 0x0" ::: "memory");
}

// ---------------- f32 -> bf16 conversion ----------------
__global__ void cvt_f32_bf16(const float* __restrict__ in,
                             unsigned short* __restrict__ out, int n4) {
  int i = blockIdx.x * blockDim.x + threadIdx.x;
  if (i >= n4) return;
  float4 f = ((const float4*)in)[i];
  ushort4 o;
  o.x = f2bf(f.x); o.y = f2bf(f.y); o.z = f2bf(f.z); o.w = f2bf(f.w);
  ((ushort4*)out)[i] = o;
}

// ---------------- generic WMMA GEMM: C[M,N] = A[M,K] * W[N,K]^T (+bias) ----------------
// Block: 128 threads (4 waves, 2x2). WG tile 128x128, wave tile 64x64 (4x4 WMMA tiles).
// Staging: async global->LDS, double-buffered, one barrier per K-step.
template <bool OUT_F32>
__global__ __launch_bounds__(128) void gemm_bf16_wmma(
    const unsigned short* __restrict__ A,   // [M,K] bf16
    const unsigned short* __restrict__ W,   // [N,K] bf16
    const float* __restrict__ bias,         // [N] or nullptr
    void* __restrict__ C,                   // [M,N] f32 or bf16
    int M, int N, int K) {
  __shared__ alignas(16) unsigned short sA[2][128][40];  // 32 cols + 8 pad
  __shared__ alignas(16) unsigned short sB[2][128][40];

  const int tid  = threadIdx.x;
  const int wave = tid >> 5, lane = tid & 31;
  const int half = lane >> 4, l16 = lane & 15;
  const int wm = wave >> 1, wn = wave & 1;            // 2x2 wave grid
  const int row0 = blockIdx.y * 128, col0 = blockIdx.x * 128;
  const int sr = tid >> 2, sc = (tid & 3) * 8;        // staging coords (32 rows/pass)

  f32x8 acc[4][4] = {};

  auto stage_async = [&](int buf, int k0) {
#pragma unroll
    for (int p = 0; p < 4; ++p) {
      async_load_b128((unsigned)(uintptr_t)&sA[buf][sr + 32 * p][sc],
                      &A[(size_t)(row0 + sr + 32 * p) * K + k0 + sc]);
      async_load_b128((unsigned)(uintptr_t)&sB[buf][sr + 32 * p][sc],
                      &W[(size_t)(col0 + sr + 32 * p) * K + k0 + sc]);
    }
  };

  stage_async(0, 0);
  int cur = 0;
  for (int k0 = 0; k0 < K; k0 += 32) {
    wait_async0();                                     // my tile-t async writes landed
    __syncthreads();                                   // everyone's landed; prior reads done
    if (k0 + 32 < K) {
      stage_async(cur ^ 1, k0 + 32);                   // DMA next tile behind compute
      if (k0 + 64 < K) {                               // warm L2 two tiles ahead
        __builtin_prefetch(&A[(size_t)(row0 + sr) * K + k0 + 64 + sc], 0, 1);
        __builtin_prefetch(&W[(size_t)(col0 + sr) * K + k0 + 64 + sc], 0, 1);
      }
    }

    FragA a[4], b;
#pragma unroll
    for (int i = 0; i < 4; ++i) {
      const int r = wm * 64 + i * 16 + l16;
      a[i].q[0] = *(const uint4*)&sA[cur][r][8 * half];     // A: k {8h..}, {16+8h..}
      a[i].q[1] = *(const uint4*)&sA[cur][r][16 + 8 * half];
    }
#pragma unroll
    for (int j = 0; j < 4; ++j) {
      const int c = wn * 64 + j * 16 + l16;
      b.q[0] = *(const uint4*)&sB[cur][c][8 * half];        // B: k {8h..}, {16+8h..}
      b.q[1] = *(const uint4*)&sB[cur][c][16 + 8 * half];
#pragma unroll
      for (int i = 0; i < 4; ++i)
        acc[i][j] = wmma_bf16(a[i].v, b.v, acc[i][j]);
    }
    cur ^= 1;
  }

  // epilogue: D element (e,half,l16) -> m = e + 8*half, n = l16
#pragma unroll
  for (int i = 0; i < 4; ++i)
#pragma unroll
    for (int j = 0; j < 4; ++j)
#pragma unroll
      for (int e = 0; e < 8; ++e) {
        const int gm = row0 + wm * 64 + i * 16 + e + 8 * half;
        const int gn = col0 + wn * 64 + j * 16 + l16;
        float v = acc[i][j][e];
        if (bias) v += bias[gn];
        if (OUT_F32) ((float*)C)[(size_t)gm * N + gn] = v;
        else         ((unsigned short*)C)[(size_t)gm * N + gn] = f2bf(v);
      }
}

// ---------------- RoPE (interleaved pairs) on bf16 Q and K ----------------
__global__ void rope_kernel(unsigned short* __restrict__ Q,
                            unsigned short* __restrict__ Kk) {
  const int total = B_ * S_ * (H_ + HKV_) * (HD_ / 2);
  int idx = blockIdx.x * blockDim.x + threadIdx.x;
  if (idx >= total) return;
  const int hp   = idx & 63;           // pair index 0..63
  const int t    = idx >> 6;
  const int head = t % (H_ + HKV_);
  const int bs   = t / (H_ + HKV_);    // b*S + s
  const int s    = bs % S_;

  const float theta = powf(1.0e6f, -(float)(2 * hp) * (1.0f / (float)HD_));
  const float fr = (float)s * theta;
  const float c = cosf(fr), sn = sinf(fr);

  unsigned short* p;
  if (head < H_) p = Q  + ((size_t)bs * QDIM  + (size_t)head * HD_ + 2 * hp);
  else           p = Kk + ((size_t)bs * KVDIM + (size_t)(head - H_) * HD_ + 2 * hp);
  const float x1 = bf2f(p[0]), x2 = bf2f(p[1]);
  p[0] = f2bf(x1 * c - x2 * sn);
  p[1] = f2bf(x1 * sn + x2 * c);
}

// ---------------- causal flash attention, GQA 32q/8kv heads ----------------
// Grid: (S/64, H, B). Block: 128 threads (4 waves), each wave owns 16 q rows.
// K/V chunk loads are register-pipelined one iteration ahead of compute.
__global__ __launch_bounds__(128) void attn_kernel(
    const unsigned short* __restrict__ Qg,   // [B,S,H,HD] bf16 (rope'd)
    const unsigned short* __restrict__ Kg,   // [B,S,HKV,HD] bf16 (rope'd)
    const unsigned short* __restrict__ Vg,   // [B,S,HKV,HD] bf16
    unsigned short* __restrict__ Og) {       // [B,S,H,HD] bf16
  __shared__ alignas(16) unsigned short sK[32][136];      // key-major K chunk
  __shared__ alignas(16) unsigned short sVt[128][40];     // V^T (hd-major)
  __shared__ alignas(16) unsigned short sP[4][16][40];    // per-wave P tile

  const int b = blockIdx.z, h = blockIdx.y, qt = blockIdx.x;
  const int kvh = h >> 2;                                 // H/HKV = 4
  const int q0 = qt * 64;
  const int tid = threadIdx.x;
  const int wave = tid >> 5, lane = tid & 31;
  const int half = lane >> 4, l16 = lane & 15;

  // Q fragments held in registers for all 4 hd-chunks (A operand)
  FragA qa[4];
  {
    const size_t qb = ((size_t)(b * S_ + q0 + wave * 16 + l16)) * QDIM + (size_t)h * HD_;
#pragma unroll
    for (int kt = 0; kt < 4; ++kt) {
      qa[kt].q[0] = *(const uint4*)&Qg[qb + kt * 32 + 8 * half];
      qa[kt].q[1] = *(const uint4*)&Qg[qb + kt * 32 + 16 + 8 * half];
    }
  }

  f32x8 Oacc[8] = {};
  float m_i[8], l_i[8];
#pragma unroll
  for (int e = 0; e < 8; ++e) { m_i[e] = -__builtin_inff(); l_i[e] = 0.0f; }

  const int lr  = tid >> 2;           // 0..31 : key row for staging
  const int lc0 = (tid & 3) * 32;     // hd column block
  const int kmax = q0 + 64;           // uniform causal bound for the WG

  uint4 rk[4], rv[4];
  {
    const size_t kb = ((size_t)(b * S_ + lr)) * KVDIM + (size_t)kvh * HD_ + lc0;
#pragma unroll
    for (int p = 0; p < 4; ++p) rk[p] = *(const uint4*)&Kg[kb + p * 8];
#pragma unroll
    for (int p = 0; p < 4; ++p) rv[p] = *(const uint4*)&Vg[kb + p * 8];
  }

  for (int kc = 0; kc < kmax; kc += 32) {
    __syncthreads();   // previous iteration's LDS reads complete
    {
#pragma unroll
      for (int p = 0; p < 4; ++p)
        *(uint4*)&sK[lr][lc0 + p * 8] = rk[p];
      union { uint4 q[4]; unsigned short u[32]; } vt;
#pragma unroll
      for (int p = 0; p < 4; ++p) vt.q[p] = rv[p];
#pragma unroll
      for (int e = 0; e < 32; ++e)
        sVt[lc0 + e][lr] = vt.u[e];   // transpose into hd-major
    }
    __syncthreads();

    // issue next chunk's global loads; they fly behind WMMA + softmax
    if (kc + 32 < kmax) {
      const size_t kb = ((size_t)(b * S_ + kc + 32 + lr)) * KVDIM + (size_t)kvh * HD_ + lc0;
#pragma unroll
      for (int p = 0; p < 4; ++p) rk[p] = *(const uint4*)&Kg[kb + p * 8];
#pragma unroll
      for (int p = 0; p < 4; ++p) rv[p] = *(const uint4*)&Vg[kb + p * 8];
      if (kc + 64 < kmax) {
        const size_t kb2 = ((size_t)(b * S_ + kc + 64 + lr)) * KVDIM + (size_t)kvh * HD_ + lc0;
        __builtin_prefetch(&Kg[kb2], 0, 1);
        __builtin_prefetch(&Vg[kb2], 0, 1);
      }
    }

    // S = Q * K^T   (2 key tiles x 4 hd chunks)
    f32x8 sacc[2] = {};
#pragma unroll
    for (int nt = 0; nt < 2; ++nt) {
      const int key = nt * 16 + l16;
#pragma unroll
      for (int kt = 0; kt < 4; ++kt) {
        FragA kb;
        kb.q[0] = *(const uint4*)&sK[key][kt * 32 + 8 * half];
        kb.q[1] = *(const uint4*)&sK[key][kt * 32 + 16 + 8 * half];
        sacc[nt] = wmma_bf16(qa[kt].v, kb.v, sacc[nt]);
      }
    }

    // online softmax (row stats shared by the 16-lane group of each half)
#pragma unroll
    for (int e = 0; e < 8; ++e) {
      const int grow = q0 + wave * 16 + e + 8 * half;
      float s0 = sacc[0][e] * SCALE_;
      float s1 = sacc[1][e] * SCALE_;
      if (kc + l16 > grow)      s0 = -__builtin_inff();
      if (kc + 16 + l16 > grow) s1 = -__builtin_inff();
      const float mx = row16_max(fmaxf(s0, s1));
      const float nm = fmaxf(m_i[e], mx);
      const float alpha = __expf(m_i[e] - nm);
      const float p0 = __expf(s0 - nm);
      const float p1 = __expf(s1 - nm);
      const float rs = row16_sum(p0 + p1);
      l_i[e] = l_i[e] * alpha + rs;
      m_i[e] = nm;
#pragma unroll
      for (int nt2 = 0; nt2 < 8; ++nt2) Oacc[nt2][e] = Oacc[nt2][e] * alpha;
      const int m = e + 8 * half;
      sP[wave][m][l16]      = f2bf(p0);
      sP[wave][m][16 + l16] = f2bf(p1);
    }
    __syncthreads();   // P visible wave-wide before A-layout reload

    // O += P * V   (P is one 16x32 A fragment, 8 hd tiles)
    FragA pa;
    pa.q[0] = *(const uint4*)&sP[wave][l16][8 * half];
    pa.q[1] = *(const uint4*)&sP[wave][l16][16 + 8 * half];
#pragma unroll
    for (int nt2 = 0; nt2 < 8; ++nt2) {
      FragA vb;
      const int hdc = nt2 * 16 + l16;
      vb.q[0] = *(const uint4*)&sVt[hdc][8 * half];
      vb.q[1] = *(const uint4*)&sVt[hdc][16 + 8 * half];
      Oacc[nt2] = wmma_bf16(pa.v, vb.v, Oacc[nt2]);
    }
  }

  // normalize and store attention output (bf16)
#pragma unroll
  for (int e = 0; e < 8; ++e) {
    const float inv = 1.0f / l_i[e];
    const int grow = q0 + wave * 16 + e + 8 * half;
    const size_t ob = ((size_t)(b * S_ + grow)) * QDIM + (size_t)h * HD_;
#pragma unroll
    for (int nt2 = 0; nt2 < 8; ++nt2)
      Og[ob + nt2 * 16 + l16] = f2bf(Oacc[nt2][e] * inv);
  }
}

// ---------------- host orchestration ----------------
extern "C" void kernel_launch(void* const* d_in, const int* in_sizes, int n_in,
                              void* d_out, int out_size, void* d_ws, size_t ws_size,
                              hipStream_t stream) {
  (void)in_sizes; (void)n_in; (void)out_size; (void)ws_size;
  const float* x  = (const float*)d_in[0];
  // d_in[1] = mask (causal; implemented analytically)
  const float* wq = (const float*)d_in[2];
  const float* wk = (const float*)d_in[3];
  const float* bk = (const float*)d_in[4];
  const float* wv = (const float*)d_in[5];
  const float* wo = (const float*)d_in[6];

  unsigned short* p = (unsigned short*)d_ws;
  unsigned short* xb  = p; p += (size_t)NTOK  * DMODEL;
  unsigned short* wqb = p; p += (size_t)QDIM  * DMODEL;
  unsigned short* wkb = p; p += (size_t)KVDIM * DMODEL;
  unsigned short* wvb = p; p += (size_t)KVDIM * DMODEL;
  unsigned short* wob = p; p += (size_t)DMODEL * QDIM;
  unsigned short* Qb  = p; p += (size_t)NTOK * QDIM;
  unsigned short* Kb  = p; p += (size_t)NTOK * KVDIM;
  unsigned short* Vb  = p; p += (size_t)NTOK * KVDIM;
  unsigned short* Ab  = p; p += (size_t)NTOK * QDIM;

  auto cvt = [&](const float* in, unsigned short* outp, size_t n) {
    int n4 = (int)(n / 4);
    cvt_f32_bf16<<<(n4 + 255) / 256, 256, 0, stream>>>(in, outp, n4);
  };
  cvt(x,  xb,  (size_t)NTOK  * DMODEL);
  cvt(wq, wqb, (size_t)QDIM  * DMODEL);
  cvt(wk, wkb, (size_t)KVDIM * DMODEL);
  cvt(wv, wvb, (size_t)KVDIM * DMODEL);
  cvt(wo, wob, (size_t)DMODEL * QDIM);

  // projections
  gemm_bf16_wmma<false><<<dim3(QDIM / 128,  NTOK / 128), 128, 0, stream>>>(
      xb, wqb, nullptr, Qb, NTOK, QDIM, DMODEL);
  gemm_bf16_wmma<false><<<dim3(KVDIM / 128, NTOK / 128), 128, 0, stream>>>(
      xb, wkb, bk, Kb, NTOK, KVDIM, DMODEL);
  gemm_bf16_wmma<false><<<dim3(KVDIM / 128, NTOK / 128), 128, 0, stream>>>(
      xb, wvb, nullptr, Vb, NTOK, KVDIM, DMODEL);

  // RoPE on Q and K
  {
    const int total = B_ * S_ * (H_ + HKV_) * (HD_ / 2);
    rope_kernel<<<(total + 255) / 256, 256, 0, stream>>>(Qb, Kb);
  }

  // causal flash attention
  attn_kernel<<<dim3(S_ / 64, H_, B_), 128, 0, stream>>>(Qb, Kb, Vb, Ab);

  // output projection (f32 straight into d_out)
  gemm_bf16_wmma<true><<<dim3(DMODEL / 128, NTOK / 128), 128, 0, stream>>>(
      Ab, wob, nullptr, (float*)d_out, NTOK, DMODEL, QDIM);
}
